// Stage5_56212531970209
// MI455X (gfx1250) — compile-verified, matrix operand order checked by reading, not run on hardware
//
#include <hip/hip_runtime.h>
#include <hip/hip_bf16.h>
#include <math.h>

// Problem constants (from reference)
#define TQ   32
#define TK   48
#define BB   64
#define HH   1024
#define G4   (4*HH)
#define MASK_FILL (-65504.0f)

typedef __attribute__((ext_vector_type(16))) __bf16        v16bf;
typedef __attribute__((ext_vector_type(8)))  float         v8f;
typedef __attribute__((ext_vector_type(8)))  unsigned int  v8u;
typedef unsigned short u16;

__device__ __forceinline__ u16 f2bf(float f) {
    unsigned u = __float_as_uint(f);
    u += 0x7FFFu + ((u >> 16) & 1u);            // round-to-nearest-even
    return (u16)(u >> 16);
}

// ---------------- generic f32 -> bf16 convert ----------------
__global__ void f32_to_bf16_kernel(const float* __restrict__ src,
                                   u16* __restrict__ dst, int n) {
    int i = blockIdx.x * blockDim.x + threadIdx.x;
    if (i < n) dst[i] = f2bf(src[i]);
}

// ---------------- vhat = v / ||v|| * norm_scalar ----------------
__global__ void vhat_kernel(const float* __restrict__ v,
                            const float* __restrict__ nscal,
                            float* __restrict__ vhat) {
    __shared__ float red[256];
    __shared__ float inv;
    int tid = threadIdx.x;
    float s = 0.f;
    for (int h = tid; h < HH; h += 256) { float x = v[h]; s += x * x; }
    red[tid] = s; __syncthreads();
    for (int o = 128; o > 0; o >>= 1) {
        if (tid < o) red[tid] += red[tid + o];
        __syncthreads();
    }
    if (tid == 0) inv = nscal[0] * rsqrtf(red[0]);
    __syncthreads();
    for (int h = tid; h < HH; h += 256) vhat[h] = v[h] * inv;
}

// ---------------- embedding gather -> bf16 ----------------
__global__ void embed_kernel(const int* __restrict__ tok,
                             const float* __restrict__ emb,
                             u16* __restrict__ xbf) {
    int idx = blockIdx.x * blockDim.x + threadIdx.x;   // TQ*BB*HH
    int row = idx >> 10;            // (t*BB + b)
    int h   = idx & (HH - 1);
    int t   = tok[row];
    xbf[idx] = f2bf(emb[(size_t)t * HH + h]);
}

// ---------------- ctx = in0 + in1 (f32 + bf16) ----------------
__global__ void ctxadd_kernel(const float* __restrict__ a,
                              const float* __restrict__ b,
                              float* __restrict__ cf,
                              u16* __restrict__ cb, int n) {
    int i = blockIdx.x * blockDim.x + threadIdx.x;
    if (i < n) { float s = a[i] + b[i]; cf[i] = s; cb[i] = f2bf(s); }
}

// ---------------- zero LSTM state ----------------
__global__ void init_state_kernel(float* __restrict__ c, u16* __restrict__ h0) {
    int i = blockIdx.x * blockDim.x + threadIdx.x;     // BB*HH
    c[i] = 0.f; h0[i] = 0;
}

// ---------------- WMMA bf16 GEMM: C = A(MxK) * W(NxK)^T + init/bias ----------------
// Register-blocked: one wave computes a (16*MT) x (16*NT) block of C.
// MT A-fragments and NT B-fragments feed MT*NT v_wmma_f32_16x16x32_bf16 per K-step,
// cutting L2 fragment traffic by ~ (MT*NT)/( (MT+NT)/2 ) vs 1 tile/wave.
// Fragment layout per CDNA5 ISA 7.12.2 (16-bit A 16x32): lane = kHalf*16 + m,
// element pair j -> K = (j<4 ? kHalf*8 + 2j : 16 + kHalf*8 + 2(j-4)).
template <int MT, int NT>
__global__ __launch_bounds__(256) void gemm_bf16_wmma(
    const u16* __restrict__ A, int lda,
    const u16* __restrict__ W, int ldw,
    float* __restrict__ C, int ldc,
    const float* __restrict__ initC, int ldinit,
    const float* __restrict__ bias0, const float* __restrict__ bias1,
    int bM, int bN, int K) {

    int wave = (blockIdx.x * blockDim.x + threadIdx.x) >> 5;
    if (wave >= bM * bN) return;                    // wave-uniform exit (EXEC stays full)
    int bm = wave / bN, bn = wave % bN;
    int m0 = bm * 16 * MT, n0 = bn * 16 * NT;
    int lane = threadIdx.x & 31;
    int half = lane >> 4, l15 = lane & 15;
    int kh8 = half * 8;

    v8f acc[MT][NT];
    #pragma unroll
    for (int mi = 0; mi < MT; ++mi) {
        #pragma unroll
        for (int ni = 0; ni < NT; ++ni) {
            if (initC) {
                #pragma unroll
                for (int r = 0; r < 8; ++r)
                    acc[mi][ni][r] = initC[(size_t)(m0 + mi * 16 + r + half * 8) * ldinit
                                           + n0 + ni * 16 + l15];
            } else if (bias0) {
                float bv = bias0[n0 + ni * 16 + l15]
                         + (bias1 ? bias1[n0 + ni * 16 + l15] : 0.f);
                #pragma unroll
                for (int r = 0; r < 8; ++r) acc[mi][ni][r] = bv;
            } else {
                #pragma unroll
                for (int r = 0; r < 8; ++r) acc[mi][ni][r] = 0.f;
            }
        }
    }

    const u16* Arow[MT];
    const u16* Wrow[NT];
    #pragma unroll
    for (int mi = 0; mi < MT; ++mi) Arow[mi] = A + (size_t)(m0 + mi * 16 + l15) * lda;
    #pragma unroll
    for (int ni = 0; ni < NT; ++ni) Wrow[ni] = W + (size_t)(n0 + ni * 16 + l15) * ldw;

    for (int k0 = 0; k0 < K; k0 += 32) {
        v8u au[MT], bu[NT];
        #pragma unroll
        for (int mi = 0; mi < MT; ++mi) {
            #pragma unroll
            for (int j = 0; j < 8; ++j) {
                int kb = k0 + ((j < 4) ? (kh8 + 2 * j) : (16 + kh8 + 2 * (j - 4)));
                au[mi][j] = *(const unsigned int*)(Arow[mi] + kb);
            }
        }
        #pragma unroll
        for (int ni = 0; ni < NT; ++ni) {
            #pragma unroll
            for (int j = 0; j < 8; ++j) {
                int kb = k0 + ((j < 4) ? (kh8 + 2 * j) : (16 + kh8 + 2 * (j - 4)));
                bu[ni][j] = *(const unsigned int*)(Wrow[ni] + kb);
            }
        }
        #pragma unroll
        for (int mi = 0; mi < MT; ++mi) {
            #pragma unroll
            for (int ni = 0; ni < NT; ++ni) {
                acc[mi][ni] = __builtin_amdgcn_wmma_f32_16x16x32_bf16(
                    false, __builtin_bit_cast(v16bf, au[mi]),
                    false, __builtin_bit_cast(v16bf, bu[ni]),
                    (short)0, acc[mi][ni], false, false);
            }
        }
    }

    #pragma unroll
    for (int mi = 0; mi < MT; ++mi) {
        #pragma unroll
        for (int ni = 0; ni < NT; ++ni) {
            #pragma unroll
            for (int r = 0; r < 8; ++r)
                C[(size_t)(m0 + mi * 16 + r + half * 8) * ldc + n0 + ni * 16 + l15]
                    = acc[mi][ni][r];
        }
    }
}

// ---------------- LSTM pointwise (one time step) ----------------
__global__ void lstm_point_kernel(const float* __restrict__ gates,
                                  float* __restrict__ c,
                                  float* __restrict__ hf,
                                  u16* __restrict__ hb) {
    int idx = blockIdx.x * blockDim.x + threadIdx.x;   // BB*HH
    int b = idx >> 10;
    int h = idx & (HH - 1);
    const float* g = gates + (size_t)b * G4;
    float gi = g[h], gf = g[HH + h], gg = g[2 * HH + h], go = g[3 * HH + h];
    float i_ = 1.f / (1.f + __expf(-gi));
    float f_ = 1.f / (1.f + __expf(-gf));
    float o_ = 1.f / (1.f + __expf(-go));
    float cv = f_ * c[idx] + i_ * tanhf(gg);
    c[idx] = cv;
    float hv = o_ * tanhf(cv);
    hf[idx] = hv;
    hb[idx] = f2bf(hv);
}

// ---------------- attention: scores -> masked softmax -> context ----------------
// one block per (q,b): blockIdx.x = q*BB + b
__global__ __launch_bounds__(256) void attn_kernel(
    const float* __restrict__ aq, const float* __restrict__ ak,
    const float* __restrict__ vhat, const float* __restrict__ nbias,
    const float* __restrict__ ctx, const int* __restrict__ lens,
    float* __restrict__ out_ctx) {

    __shared__ float aqb[HH];
    __shared__ float vh[HH];
    __shared__ float sc[64];

    int row = blockIdx.x;          // q*BB + b
    int b = row & (BB - 1);
    int tid = threadIdx.x;

    // Async-copy vhat (4KB) into LDS: one 16B ASYNCcnt-tracked transfer per thread.
    // Generic LDS pointers carry the wave-relative LDS offset in their low 32 bits.
    {
        unsigned ldsa = (unsigned)(uintptr_t)(void*)vh + (unsigned)(tid * 16);
        const float* gsrc = vhat + tid * 4;
        asm volatile("global_load_async_to_lds_b128 %0, %1, off"
                     :: "v"(ldsa), "v"(gsrc) : "memory");
    }
    for (int h = tid; h < HH; h += 256)
        aqb[h] = aq[(size_t)row * HH + h] + nbias[h];
    asm volatile("s_wait_asynccnt 0x0" ::: "memory");
    __syncthreads();

    int wave = tid >> 5, lane = tid & 31;
    for (int k = wave; k < TK; k += 8) {
        const float* akr = ak + ((size_t)k * BB + b) * HH;
        if (k + 8 < TK) {   // prefetch next ak row: one 128B line per lane covers 4KB
            const char* nxt = (const char*)(ak + ((size_t)(k + 8) * BB + b) * HH);
            __builtin_prefetch(nxt + lane * 128, 0, 0);
        }
        float s = 0.f;
        for (int h = lane; h < HH; h += 32)
            s += vh[h] * tanhf(aqb[h] + akr[h]);
        #pragma unroll
        for (int off = 16; off > 0; off >>= 1)
            s += __shfl_xor(s, off, 32);
        if (lane == 0) sc[k] = s;
    }
    __syncthreads();

    if (tid == 0) {
        int len = lens[b];
        float mx = -INFINITY;
        for (int k = 0; k < TK; ++k) {
            float v = (k < len) ? sc[k] : MASK_FILL;
            sc[k] = v;
            mx = fmaxf(mx, v);
        }
        float sum = 0.f;
        for (int k = 0; k < TK; ++k) { float e = __expf(sc[k] - mx); sc[k] = e; sum += e; }
        float inv = 1.f / sum;
        for (int k = 0; k < TK; ++k) sc[k] *= inv;
    }
    __syncthreads();

    for (int h = tid; h < HH; h += 256) {
        float acc = 0.f;
        for (int k = 0; k < TK; ++k)
            acc += sc[k] * ctx[((size_t)k * BB + b) * HH + h];
        out_ctx[(size_t)row * HH + h] = acc;
    }
}

extern "C" void kernel_launch(void* const* d_in, const int* in_sizes, int n_in,
                              void* d_out, int out_size, void* d_ws, size_t ws_size,
                              hipStream_t stream) {
    const float* in1   = (const float*)d_in[0];   // input1 (TK,B,H)
    const float* in0   = (const float*)d_in[1];   // input0 (TK,B,H)
    const int*   lens  = (const int*)  d_in[2];   // (B,)
    const int*   toks  = (const int*)  d_in[3];   // (TQ,B)
    const float* emb   = (const float*)d_in[4];   // (V,H)
    const float* Wih   = (const float*)d_in[5];   // (4H,H)
    const float* Whh   = (const float*)d_in[6];   // (4H,H)
    const float* bih   = (const float*)d_in[7];   // (4H,)
    const float* bhh   = (const float*)d_in[8];   // (4H,)
    const float* Wq    = (const float*)d_in[9];   // (H,H)
    const float* Wk    = (const float*)d_in[10];  // (H,H)
    const float* vatt  = (const float*)d_in[11];  // (H,)
    const float* nsc   = (const float*)d_in[12];  // (1,)
    const float* nbias = (const float*)d_in[13];  // (H,)

    float* out_rnn = (float*)d_out;                            // (TQ,B,H)
    float* out_ctx = out_rnn + (size_t)TQ * BB * HH;           // (TQ,B,H)

    // ---- workspace carve-up (≈103 MB) ----
    char* ws = (char*)d_ws;
    size_t off = 0;
    auto carve = [&](size_t bytes) -> char* {
        char* p = ws + off;
        off = (off + bytes + 255) & ~(size_t)255;
        return p;
    };
    u16*   Wih_bf = (u16*)  carve((size_t)G4 * HH * 2);        // 8 MB
    u16*   Whh_bf = (u16*)  carve((size_t)G4 * HH * 2);        // 8 MB
    u16*   Wq_bf  = (u16*)  carve((size_t)HH * HH * 2);        // 2 MB
    u16*   Wk_bf  = (u16*)  carve((size_t)HH * HH * 2);        // 2 MB
    u16*   x_bf   = (u16*)  carve((size_t)TQ * BB * HH * 2);   // 4 MB
    float* ctx_f  = (float*)carve((size_t)TK * BB * HH * 4);   // 12.6 MB
    u16*   ctx_b  = (u16*)  carve((size_t)TK * BB * HH * 2);   // 6 MB
    float* xp     = (float*)carve((size_t)TQ * BB * G4 * 4);   // 33.6 MB
    float* gates  = (float*)carve((size_t)BB * G4 * 4);        // 1 MB
    float* cstate = (float*)carve((size_t)BB * HH * 4);        // 256 KB
    u16*   h0_bf  = (u16*)  carve((size_t)BB * HH * 2);        // 128 KB
    u16*   rnn_bf = (u16*)  carve((size_t)TQ * BB * HH * 2);   // 4 MB
    float* aq     = (float*)carve((size_t)TQ * BB * HH * 4);   // 8.4 MB
    float* ak     = (float*)carve((size_t)TK * BB * HH * 4);   // 12.6 MB
    float* vhat   = (float*)carve((size_t)HH * 4);
    (void)ws_size; (void)in_sizes; (void)n_in; (void)out_size;

    // ---- 1. weight conversions + vhat ----
    f32_to_bf16_kernel<<<(G4 * HH) / 256, 256, 0, stream>>>(Wih, Wih_bf, G4 * HH);
    f32_to_bf16_kernel<<<(G4 * HH) / 256, 256, 0, stream>>>(Whh, Whh_bf, G4 * HH);
    f32_to_bf16_kernel<<<(HH * HH) / 256, 256, 0, stream>>>(Wq, Wq_bf, HH * HH);
    f32_to_bf16_kernel<<<(HH * HH) / 256, 256, 0, stream>>>(Wk, Wk_bf, HH * HH);
    vhat_kernel<<<1, 256, 0, stream>>>(vatt, nsc, vhat);

    // ---- 2. embedding gather, ctx add, state init ----
    embed_kernel<<<(TQ * BB * HH) / 256, 256, 0, stream>>>(toks, emb, x_bf);
    ctxadd_kernel<<<(TK * BB * HH) / 256, 256, 0, stream>>>(in0, in1, ctx_f, ctx_b,
                                                            TK * BB * HH);
    init_state_kernel<<<(BB * HH) / 256, 256, 0, stream>>>(cstate, h0_bf);

    // ---- 3. xp = x @ W_ih^T + b_ih + b_hh   (2048 x 4096, K=1024), 32x64/wave ----
    {
        int bM = (TQ * BB) / 32, bN = G4 / 64;                 // 64 x 64 blocks
        int blocks = (bM * bN + 7) / 8;
        gemm_bf16_wmma<2, 4><<<blocks, 256, 0, stream>>>(
            x_bf, HH, Wih_bf, HH, xp, G4,
            nullptr, 0, bih, bhh, bM, bN, HH);
    }

    // ---- 4. LSTM recurrence: 32 sequential steps (16x32/wave: keep parallelism) ----
    for (int t = 0; t < TQ; ++t) {
        const u16* hprev = (t == 0) ? h0_bf : (rnn_bf + (size_t)(t - 1) * BB * HH);
        int bM = BB / 16, bN = G4 / 32;                        // 4 x 128 blocks
        int blocks = (bM * bN + 7) / 8;
        gemm_bf16_wmma<1, 2><<<blocks, 256, 0, stream>>>(
            hprev, HH, Whh_bf, HH, gates, G4,
            xp + (size_t)t * BB * G4, G4, nullptr, nullptr, bM, bN, HH);
        lstm_point_kernel<<<(BB * HH) / 256, 256, 0, stream>>>(
            gates, cstate, out_rnn + (size_t)t * BB * HH,
            rnn_bf + (size_t)t * BB * HH);
    }

    // ---- 5. attention projections (32x64/wave) ----
    {
        int bM = (TQ * BB) / 32, bN = HH / 64;                 // 64 x 16 blocks
        int blocks = (bM * bN + 7) / 8;
        gemm_bf16_wmma<2, 4><<<blocks, 256, 0, stream>>>(
            rnn_bf, HH, Wq_bf, HH, aq, HH,
            nullptr, 0, nullptr, nullptr, bM, bN, HH);
    }
    {
        int bM = (TK * BB) / 32, bN = HH / 64;                 // 96 x 16 blocks
        int blocks = (bM * bN + 7) / 8;
        gemm_bf16_wmma<2, 4><<<blocks, 256, 0, stream>>>(
            ctx_b, HH, Wk_bf, HH, ak, HH,
            nullptr, 0, nullptr, nullptr, bM, bN, HH);
    }

    // ---- 6. scores + masked softmax + context ----
    attn_kernel<<<TQ * BB, 256, 0, stream>>>(aq, ak, vhat, nbias, ctx_f, lens, out_ctx);
}
